// PFM_17377437680151
// MI455X (gfx1250) — matrix operand order repaired and checked by reading, not compiled, not run on hardware
//
#include <hip/hip_runtime.h>
#include <hip/hip_bf16.h>

typedef __bf16 bf16_t;
typedef bf16_t v16bf __attribute__((ext_vector_type(16)));
typedef bf16_t v8bf  __attribute__((ext_vector_type(8)));
typedef bf16_t v4bf  __attribute__((ext_vector_type(4)));
typedef float  v8f   __attribute__((ext_vector_type(8)));

#define B_    16
#define T_    2048
#define H_    512
#define M_    96
#define V2_   625
#define NROW  (B_ * M_)      // 1536
#define NTILE_V 40           // 625 padded to 640 -> 40 tiles of 16
#define JTILES  6            // 96 rows = 6 j-tiles, all staged in LDS

// Dynamic LDS: qs (2 KB) + A' 96x512 bf16 (96 KB) = 100352 bytes.
// CDNA5 WGP has 320 KB LDS -> 3 blocks/WGP co-resident. (Impossible on CDNA4.)
#define PAIR_LDS_BYTES (2048 + M_ * H_ * 2)

// ---------------------------------------------------------------------------
// Kernel 0: convert W_out (f32, [V2,H]) -> bf16 for WMMA B operands
// ---------------------------------------------------------------------------
__global__ void wcvt_kernel(const float* __restrict__ W, bf16_t* __restrict__ Wb, int n) {
    int i = blockIdx.x * blockDim.x + threadIdx.x;
    if (i < n) Wb[i] = (bf16_t)W[i];
}

// ---------------------------------------------------------------------------
// Kernel 1: fused gather + projection GEMM (q and k), one wave per 16x16 tile.
//   q[n,o] = sum_h x[b, mask_idx[b,m], h] * Wq[o,h]   (n = b*M+m)
// ---------------------------------------------------------------------------
__global__ __launch_bounds__(32)
void proj_kernel(const float* __restrict__ x, const int* __restrict__ midx,
                 const float* __restrict__ Wq, const float* __restrict__ Wk,
                 float* __restrict__ qout, float* __restrict__ kout) {
    const int lane = threadIdx.x & 31;
    const int tm = blockIdx.x;                 // row tile over NROW (96 tiles)
    const int to = blockIdx.y;                 // col tile over H    (32 tiles)
    const float* W = blockIdx.z ? Wk : Wq;
    float* out     = blockIdx.z ? kout : qout;

    const int r  = lane & 15;
    const int hi = (lane >> 4) & 1;

    const int n = tm * 16 + r;
    const int b = n / M_;
    const int m = n % M_;
    const int t = midx[b * M_ + m];
    const float* xrow = x + ((size_t)b * T_ + t) * H_;

    const int o = to * 16 + r;
    const float* wrow = W + (size_t)o * H_;

    v8f acc = {};
    for (int kk = 0; kk < H_; kk += 32) {
        // A fragment: contiguous k order (k = 16*hi + s), matching B
        const int a1 = kk + (hi ? 16 : 0);
        const int a2 = a1 + 8;
        v16bf a;
        #pragma unroll
        for (int e = 0; e < 8; ++e) a[e]     = (bf16_t)xrow[a1 + e];
        #pragma unroll
        for (int e = 0; e < 8; ++e) a[8 + e] = (bf16_t)xrow[a2 + e];
        // B fragment: 32x16 layout -> one contiguous 16-elem run / lane
        const int bk = kk + (hi ? 16 : 0);
        v16bf bb;
        #pragma unroll
        for (int e = 0; e < 16; ++e) bb[e] = (bf16_t)wrow[bk + e];
        acc = __builtin_amdgcn_wmma_f32_16x16x32_bf16(false, a, false, bb,
                                                      (short)0, acc, false, false);
    }
    #pragma unroll
    for (int v = 0; v < 8; ++v) {
        int row = tm * 16 + v + hi * 8;
        out[(size_t)row * H_ + o] = acc[v];
    }
}

// ---------------------------------------------------------------------------
// Kernel 2: per-(b,i) rank-scaled GEMM.
//   out[sb, j, v] = sum_h (q[sb,h] * k[b*M+j,h]) * Wb[v,h]
// One workgroup (8 waves) per sb. The FULL 96x512 A' lives in LDS (96 KB),
// so each W_out B-fragment load feeds 6 WMMAs (6 accumulators per wave):
// every workgroup streams W_out from L2 exactly once (640 KB).
// ---------------------------------------------------------------------------
__global__ __launch_bounds__(256)
void pair_kernel(const float* __restrict__ qbuf, const float* __restrict__ kbuf,
                 const bf16_t* __restrict__ Wb, float* __restrict__ out) {
    extern __shared__ char smem[];
    float*  qs = reinterpret_cast<float*>(smem);            // 2 KB
    bf16_t* As = reinterpret_cast<bf16_t*>(smem + 2048);    // 96 KB: A'[96][512]

    const int sb   = blockIdx.x;             // b*M + i
    const int tid  = threadIdx.x;
    const int lane = tid & 31;
    const int wave = tid >> 5;
    const int b    = sb / M_;

    for (int h4 = tid; h4 < H_ / 4; h4 += 256)
        *reinterpret_cast<float4*>(&qs[h4 * 4]) =
            *reinterpret_cast<const float4*>(&qbuf[(size_t)sb * H_ + h4 * 4]);
    __syncthreads();

    // Build A': 96 rows x 512, f32 product rounded once to bf16
    for (int e4 = tid; e4 < M_ * H_ / 4; e4 += 256) {
        int e  = e4 * 4;
        int j  = e >> 9;                      // / H_
        int h  = e & (H_ - 1);
        float4 kv = *reinterpret_cast<const float4*>(
                        &kbuf[((size_t)(b * M_ + j)) * H_ + h]);
        float4 qv = *reinterpret_cast<const float4*>(&qs[h]);
        v4bf p;
        p[0] = (bf16_t)(qv.x * kv.x);
        p[1] = (bf16_t)(qv.y * kv.y);
        p[2] = (bf16_t)(qv.z * kv.z);
        p[3] = (bf16_t)(qv.w * kv.w);
        *reinterpret_cast<v4bf*>(&As[e]) = p;
    }
    __syncthreads();

    const int r  = lane & 15;
    const int hi = (lane >> 4) & 1;

    #pragma unroll 1
    for (int t = 0; t < NTILE_V / 8; ++t) {                  // 5 v-tiles / wave
        const int vt  = wave * (NTILE_V / 8) + t;
        const int vc  = vt * 16 + r;
        const int vcl = vc < V2_ ? vc : V2_ - 1;             // clamp, EXEC stays full
        const bf16_t* wrow = Wb + (size_t)vcl * H_ + (hi ? 16 : 0);

        v8f acc[JTILES];
        #pragma unroll
        for (int jl = 0; jl < JTILES; ++jl) acc[jl] = (v8f){};

        #pragma unroll 2
        for (int kk = 0; kk < H_; kk += 32) {
            v16bf bb = *reinterpret_cast<const v16bf*>(wrow + kk);
            const int a1 = kk + (hi ? 16 : 0);
            const int a2 = a1 + 8;
            #pragma unroll
            for (int jl = 0; jl < JTILES; ++jl) {            // 6 WMMAs per B-frag
                v8bf lo = *reinterpret_cast<const v8bf*>(&As[(jl * 16 + r) * H_ + a1]);
                v8bf hh = *reinterpret_cast<const v8bf*>(&As[(jl * 16 + r) * H_ + a2]);
                v16bf a = __builtin_shufflevector(lo, hh,
                    0,1,2,3,4,5,6,7,8,9,10,11,12,13,14,15);
                acc[jl] = __builtin_amdgcn_wmma_f32_16x16x32_bf16(
                              false, a, false, bb, (short)0, acc[jl], false, false);
            }
        }

        if (vc < V2_) {   // predicated NT stores only (after all WMMAs)
            const size_t base = (size_t)sb * M_ * V2_ + vc;
            #pragma unroll
            for (int jl = 0; jl < JTILES; ++jl) {
                #pragma unroll
                for (int v = 0; v < 8; ++v) {
                    int row = jl * 16 + v + hi * 8;
                    __builtin_nontemporal_store(acc[jl][v],
                        &out[base + (size_t)row * V2_]);
                }
            }
        }
    }
}

// ---------------------------------------------------------------------------
// Kernel 3: diag mask = eye(M) flattened, tiled over batch (second output)
// ---------------------------------------------------------------------------
__global__ void diag_kernel(float* __restrict__ dm, int n) {
    int idx = blockIdx.x * blockDim.x + threadIdx.x;
    if (idx < n) {
        int j = idx % M_;
        int i = (idx / M_) % M_;
        __builtin_nontemporal_store((i == j) ? 1.0f : 0.0f, &dm[idx]);
    }
}

extern "C" void kernel_launch(void* const* d_in, const int* in_sizes, int n_in,
                              void* d_out, int out_size, void* d_ws, size_t ws_size,
                              hipStream_t stream) {
    const float* x    = (const float*)d_in[0];
    const float* Wq   = (const float*)d_in[1];
    const float* Wk   = (const float*)d_in[2];
    const float* Wout = (const float*)d_in[3];
    const int*   midx = (const int*)d_in[4];

    // Workspace: qbuf f32 [1536,512], kbuf f32 [1536,512], Wb bf16 [625,512]
    char* ws = (char*)d_ws;
    float*  qbuf = (float*)(ws);
    float*  kbuf = (float*)(ws + (size_t)NROW * H_ * sizeof(float));
    bf16_t* Wb   = (bf16_t*)(ws + 2u * (size_t)NROW * H_ * sizeof(float));

    // Opt pair_kernel into >64 KB dynamic LDS (no-op if unnecessary; not a
    // stream operation, safe under graph capture, idempotent).
    (void)hipFuncSetAttribute((const void*)pair_kernel,
                              hipFuncAttributeMaxDynamicSharedMemorySize,
                              PAIR_LDS_BYTES);

    {
        int n = V2_ * H_;
        hipLaunchKernelGGL(wcvt_kernel, dim3((n + 255) / 256), dim3(256), 0, stream,
                           Wout, Wb, n);
    }
    hipLaunchKernelGGL(proj_kernel, dim3(NROW / 16, H_ / 16, 2), dim3(32), 0, stream,
                       x, midx, Wq, Wk, qbuf, kbuf);
    hipLaunchKernelGGL(pair_kernel, dim3(NROW), dim3(256), PAIR_LDS_BYTES, stream,
                       qbuf, kbuf, Wb, (float*)d_out);
    {
        int nd = B_ * M_ * M_;
        float* dm = (float*)d_out + (size_t)nd * V2_;
        hipLaunchKernelGGL(diag_kernel, dim3((nd + 255) / 256), dim3(256), 0, stream,
                           dm, nd);
    }
}
